// MoE_36747740184922
// MI455X (gfx1250) — matrix operand order repaired and checked by reading, not compiled, not run on hardware
//
#include <hip/hip_runtime.h>
#include <hip/hip_bf16.h>
#include <math.h>

#define B_ 512
#define S_ 512
#define I_ 128
#define H_ 128
#define E_ 8
#define F_ 20

typedef __attribute__((ext_vector_type(16))) __bf16    v16bf;
typedef __attribute__((ext_vector_type(8)))  float     v8f;
typedef __attribute__((ext_vector_type(4)))  unsigned  uiv4;

// WMMA fragment viewed as raw bf16 bit patterns / 128-bit chunks.
union AV {
    v16bf          v;
    unsigned short s[16];
    unsigned       u[8];
    uiv4           q[2];
};

__device__ __forceinline__ unsigned short f2bf(float f) {
    union { float f; unsigned u; } cv; cv.f = f;
    unsigned u = cv.u;
    u += 0x7FFFu + ((u >> 16) & 1u);           // round-to-nearest-even
    return (unsigned short)(u >> 16);
}

// Hardware transcendental tanh (CDNA5 V_TANH_F32) instead of libm expansion.
__device__ __forceinline__ float fast_tanh(float x) {
    float r;
    asm("v_tanh_f32 %0, %1" : "=v"(r) : "v"(x));
    return r;
}

// Pack two f32 -> two bf16 in one VGPR (RNE) with one instruction.
__device__ __forceinline__ unsigned pk_bf16(float a, float b) {
    unsigned r;
    asm("v_cvt_pk_bf16_f32 %0, %1, %2" : "=v"(r) : "v"(a), "v"(b));
    return r;
}

// ---------------------------------------------------------------- init ----
__global__ void moe_init_kernel(float* __restrict__ acc16) {
    if (threadIdx.x < 16) acc16[threadIdx.x] = 0.0f;
}

// --------------------------------------------------- gate + x->bf16 ----
// One pass over x: converts x to bf16 (feeds the WMMA RNN) AND computes
// x.mean(axis=1) -> logits -> top-2 softmax gates + importance/load stats.
__global__ void moe_gate_convert_kernel(const float* __restrict__ x,
                                        const float* __restrict__ w_gate,
                                        unsigned short* __restrict__ xb,
                                        float* __restrict__ gates,
                                        float* __restrict__ imp,
                                        float* __restrict__ loadv) {
    int b   = blockIdx.x;
    int tid = threadIdx.x;
    int i   = tid & 127;
    int sh  = tid >> 7;
    const float*    xp  = x  + (size_t)b * S_ * I_;
    unsigned short* xbp = xb + (size_t)b * S_ * I_;

    float sum = 0.f;
    for (int s = sh; s < S_; s += 2) {
        float v = xp[(size_t)s * I_ + i];
        sum += v;
        xbp[(size_t)s * I_ + i] = f2bf(v);
    }

    __shared__ float red[256];
    __shared__ float logits[E_];
    red[tid] = sum;
    __syncthreads();
    float mean = 0.f;
    if (tid < 128) mean = (red[tid] + red[tid + 128]) * (1.0f / S_);
    __syncthreads();

    for (int e = 0; e < E_; ++e) {
        if (tid < 128) red[tid] = mean * w_gate[i * E_ + e];
        __syncthreads();
        for (int off = 64; off > 0; off >>= 1) {
            if (tid < off) red[tid] += red[tid + off];
            __syncthreads();
        }
        if (tid == 0) logits[e] = red[0];
        __syncthreads();
    }
    if (tid == 0) {
        int i1 = 0;
        for (int e = 1; e < E_; ++e) if (logits[e] > logits[i1]) i1 = e;
        int i2 = (i1 == 0) ? 1 : 0;
        for (int e = 0; e < E_; ++e) if (e != i1 && logits[e] > logits[i2]) i2 = e;
        float v1 = logits[i1], v2 = logits[i2];
        float m  = fmaxf(v1, v2);
        float e1 = __expf(v1 - m), e2 = __expf(v2 - m);
        float inv = 1.0f / (e1 + e2);
        float g1 = e1 * inv, g2 = e2 * inv;
        for (int e = 0; e < E_; ++e) gates[b * E_ + e] = 0.f;
        gates[b * E_ + i1] = g1;
        gates[b * E_ + i2] = g2;
        atomicAdd(&imp[i1], g1);    atomicAdd(&imp[i2], g2);
        atomicAdd(&loadv[i1], 1.f); atomicAdd(&loadv[i2], 1.f);
    }
}

// ----------------------------------------------------------------- rnn ----
// One RNN timestep. AXC = staged A-fragments for step T (already in VGPRs);
// AXN = buffer to prefetch step T+1's fragments into (double buffering, so
// the x-side WMMA chain never waits on global loads). Two independent
// accumulator chains (cx, ch) break the WMMA RAW chain for ILP.
#define RNN_STEP(AXC, AXN, T)                                               \
  do {                                                                      \
    if ((T) + 1 < S_) {                                                     \
      const unsigned short* xpn = xrow + (size_t)((T) + 1) * I_;            \
      __builtin_prefetch(xpn + I_, 0, 1);  /* global_prefetch_b8: t+2 */    \
      _Pragma("unroll")                                                     \
      for (int kt = 0; kt < 4; ++kt) {                                      \
        AXN[kt].q[0] = *(const uiv4*)(xpn + kt * 32 + g * 8);               \
        AXN[kt].q[1] = *(const uiv4*)(xpn + kt * 32 + 16 + g * 8);          \
      }                                                                     \
    }                                                                       \
    AV ah[4];                                                               \
    _Pragma("unroll")                                                       \
    for (int kt = 0; kt < 4; ++kt) {                                        \
      ah[kt].q[0] = *(const uiv4*)&hbuf[m][kt * 32 + g * 8];                \
      ah[kt].q[1] = *(const uiv4*)&hbuf[m][kt * 32 + 16 + g * 8];           \
    }                                                                       \
    _Pragma("unroll")                                                       \
    for (int kt = 0; kt < 4; ++kt)                                          \
      cx = __builtin_amdgcn_wmma_f32_16x16x32_bf16(false, AXC[kt].v, false, \
                                                   Bih[kt].v, (short)0, cx, \
                                                   false, false);           \
    _Pragma("unroll")                                                       \
    for (int kt = 0; kt < 4; ++kt)                                          \
      ch = __builtin_amdgcn_wmma_f32_16x16x32_bf16(false, ah[kt].v, false,  \
                                                   Bhh[kt].v, (short)0, ch, \
                                                   false, false);           \
    __syncthreads();  /* all waves done reading hbuf */                     \
    float th[8];                                                            \
    _Pragma("unroll")                                                       \
    for (int r = 0; r < 8; ++r) th[r] = fast_tanh(cx[r] + ch[r]);           \
    _Pragma("unroll")                                                       \
    for (int r = 0; r < 8; r += 2) {                                        \
      unsigned p = pk_bf16(th[r], th[r + 1]);                               \
      hbuf[r + 8 * g][n]     = (unsigned short)(p & 0xffffu);               \
      hbuf[r + 1 + 8 * g][n] = (unsigned short)(p >> 16);                   \
    }                                                                       \
    if ((T) == S_ - 1) {                                                    \
      _Pragma("unroll")                                                     \
      for (int r = 0; r < 8; ++r)                                           \
        hT[((size_t)e * B_ + (b0 + r + 8 * g)) * H_ + n] = th[r];           \
    }                                                                       \
    _Pragma("unroll")                                                       \
    for (int r = 0; r < 8; ++r) { cx[r] = bias; ch[r] = 0.f; }              \
    __syncthreads();                                                        \
  } while (0)

// grid = E * (B/16); block = 256 threads = 8 waves.
// Wave w owns H-columns [16w, 16w+16). Weights live in registers as bf16
// WMMA B-fragments; h-state (16x128 bf16) lives in LDS; x streams in as
// pre-converted bf16 via 16-byte global_load_b128, double-buffered.
__global__ void __launch_bounds__(256, 1)
moe_rnn_wmma_kernel(const unsigned short* __restrict__ xb,
                    const float* __restrict__ W_ih,
                    const float* __restrict__ W_hh,
                    const float* __restrict__ b_ih,
                    const float* __restrict__ b_hh,
                    float* __restrict__ hT) {
    const int NB   = B_ / 16;
    const int e    = blockIdx.x / NB;
    const int b0   = (blockIdx.x % NB) * 16;
    const int wave = threadIdx.x >> 5;
    const int lane = threadIdx.x & 31;
    const int g    = lane >> 4;        // lane-half select (ISA A/B layouts)
    const int nl   = lane & 15;
    const int n    = wave * 16 + nl;   // output hidden column
    const int m    = nl;               // A-matrix row for this lane

    __shared__ unsigned short hbuf[16][H_];   // 4 KB bf16 hidden state
    for (int idx = threadIdx.x; idx < 16 * H_; idx += 256)
        ((unsigned short*)hbuf)[idx] = 0;     // h0 = 0

    // B-fragments: Bmat[k][n] = W[n][k]  (W row-major [H][I]).
    // Dense bf16 B 32x16: lane-half g holds K = kt*32 + g*8 + 0..7, kt*32 + 16 + g*8 + 0..7.
    AV Bih[4], Bhh[4];
#pragma unroll
    for (int kt = 0; kt < 4; ++kt) {
        const float* wi = W_ih + ((size_t)e * H_ + n) * I_ + kt * 32 + g * 8;
        const float* wh = W_hh + ((size_t)e * H_ + n) * H_ + kt * 32 + g * 8;
#pragma unroll
        for (int j = 0; j < 8; ++j) {
            Bih[kt].u[j] = pk_bf16(wi[2 * j + (j >= 4 ? 8 : 0)], wi[2 * j + 1 + (j >= 4 ? 8 : 0)]);
            Bhh[kt].u[j] = pk_bf16(wh[2 * j + (j >= 4 ? 8 : 0)], wh[2 * j + 1 + (j >= 4 ? 8 : 0)]);
        }
    }
    const float bias = b_ih[e * H_ + n] + b_hh[e * H_ + n];

    v8f cx, ch;
#pragma unroll
    for (int r = 0; r < 8; ++r) { cx[r] = bias; ch[r] = 0.f; }

    __syncthreads();

    const unsigned short* xrow = xb + (size_t)(b0 + m) * S_ * I_;

    // Prime the pipeline with t=0's A-fragments.
    AV ax0[4], ax1[4];
#pragma unroll
    for (int kt = 0; kt < 4; ++kt) {
        ax0[kt].q[0] = *(const uiv4*)(xrow + kt * 32 + g * 8);
        ax0[kt].q[1] = *(const uiv4*)(xrow + kt * 32 + 16 + g * 8);
    }

    for (int t = 0; t < S_; t += 2) {
        RNN_STEP(ax0, ax1, t);
        RNN_STEP(ax1, ax0, t + 1);
    }
}

// ---------------------------------------------------------------- loss ----
__global__ void moe_loss_kernel(const float* __restrict__ imp,
                                const float* __restrict__ loadv,
                                float* __restrict__ out_loss) {
    if (threadIdx.x == 0) {
        float cv[2];
        const float* arr[2] = { imp, loadv };
        for (int a = 0; a < 2; ++a) {
            float s = 0.f;
            for (int e = 0; e < E_; ++e) s += arr[a][e];
            float mean = s / E_;
            float v = 0.f;
            for (int e = 0; e < E_; ++e) { float d = arr[a][e] - mean; v += d * d; }
            v /= (float)(E_ - 1);                       // ddof=1
            cv[a] = v / (mean * mean + 1e-10f);
        }
        *out_loss = (cv[0] + cv[1]) * 0.01f;
    }
}

// ---------------------------------------------------------------- head ----
// block = batch row b; fc1 (E*F=160 dot-products over H) + fc2 + gated combine
__global__ void moe_head_kernel(const float* __restrict__ hT,
                                const float* __restrict__ fc1_w,
                                const float* __restrict__ fc1_b,
                                const float* __restrict__ fc2_w,
                                const float* __restrict__ fc2_b,
                                const float* __restrict__ gates,
                                float* __restrict__ y) {
    int b = blockIdx.x;
    int tid = threadIdx.x;
    __shared__ float zsh[E_][F_];
    __shared__ float osh[E_];
    if (tid < E_ * F_) {
        int e = tid / F_, f = tid % F_;
        const float* w = fc1_w + ((size_t)e * F_ + f) * H_;
        const float* h = hT + ((size_t)e * B_ + b) * H_;
        float acc = fc1_b[e * F_ + f];
        for (int k = 0; k < H_; ++k) acc += w[k] * h[k];
        zsh[e][f] = fast_tanh(acc);
    }
    __syncthreads();
    if (tid < E_) {
        float acc = fc2_b[tid];
        for (int f = 0; f < F_; ++f) acc += fc2_w[tid * F_ + f] * zsh[tid][f];
        osh[tid] = acc;
    }
    __syncthreads();
    if (tid == 0) {
        float acc = 0.f;
        for (int e = 0; e < E_; ++e) acc += gates[b * E_ + e] * osh[e];
        y[b] = acc;
    }
}

// -------------------------------------------------------------- launch ----
extern "C" void kernel_launch(void* const* d_in, const int* in_sizes, int n_in,
                              void* d_out, int out_size, void* d_ws, size_t ws_size,
                              hipStream_t stream) {
    const float* x      = (const float*)d_in[0];
    const float* w_gate = (const float*)d_in[1];
    const float* W_ih   = (const float*)d_in[2];
    const float* W_hh   = (const float*)d_in[3];
    const float* b_ih   = (const float*)d_in[4];
    const float* b_hh   = (const float*)d_in[5];
    const float* fc1_w  = (const float*)d_in[6];
    const float* fc1_b  = (const float*)d_in[7];
    const float* fc2_w  = (const float*)d_in[8];
    const float* fc2_b  = (const float*)d_in[9];

    float* out = (float*)d_out;            // [0..511] = y, [512] = loss

    // Workspace layout (all 16B-aligned):
    //   xb    : B*S*I bf16  (64 MB)  pre-converted activations
    //   gates : B*E f32
    //   imp   : E f32, loadv : E f32
    //   hT    : E*B*H f32   (2 MB)
    unsigned short* xb    = (unsigned short*)d_ws;
    const size_t    XBN   = (size_t)B_ * S_ * I_;
    float*          gates = (float*)((char*)d_ws + XBN * sizeof(unsigned short));
    float*          imp   = gates + (size_t)B_ * E_;
    float*          loadv = imp + E_;
    float*          hT    = loadv + E_;

    moe_init_kernel<<<1, 32, 0, stream>>>(imp);
    moe_gate_convert_kernel<<<B_, 256, 0, stream>>>(x, w_gate, xb, gates, imp, loadv);
    moe_rnn_wmma_kernel<<<E_ * (B_ / 16), 256, 0, stream>>>(xb, W_ih, W_hh,
                                                            b_ih, b_hh, hT);
    moe_loss_kernel<<<1, 32, 0, stream>>>(imp, loadv, out + B_);
    moe_head_kernel<<<B_, 256, 0, stream>>>(hT, fc1_w, fc1_b, fc2_w, fc2_b,
                                            gates, out);
}
